// FlaxAttention2_53300544143372
// MI455X (gfx1250) — matrix operand order repaired and checked by reading, not compiled, not run on hardware
//
#include <hip/hip_runtime.h>

// ---------------------------------------------------------------------------
// probs = softmax( (Q^T K over the sequence axis) * 0.125 )  per batch-head
// Q,K,V: [2, 16384, 8, 64] fp32 (V is dead).  Out: [16, 64, 64] fp32.
// Memory-bound: 128 MiB read @ 23.3 TB/s ~ 5.8us; f16 WMMA keeps compute
// off the critical path (f32 WMMA could not sustain 373 TFLOP/s equivalent).
// ---------------------------------------------------------------------------

typedef _Float16 half16 __attribute__((ext_vector_type(16)));
typedef float    float8 __attribute__((ext_vector_type(8)));

#define BH        16
#define SEQ       16384
#define DHEAD     64
#define ROWSTRIDE 512            // H * DHEAD floats between consecutive s rows
#define NCHUNK    32             // split-K chunks per batch-head
#define SCHUNK    (SEQ / NCHUNK) // 512 s-values per workgroup
#define NWAVES    8
#define SSUB      (SCHUNK / NWAVES) // 64 s-values per wave
#define KSTEPS    (SSUB / 32)       // 2 WMMA K-steps of 32 per wave

__global__ __launch_bounds__(256)
void attn_scores_wmma(const float* __restrict__ q,
                      const float* __restrict__ k,
                      float* __restrict__ ws)
{
    const int chunk = blockIdx.x;        // 0..NCHUNK-1
    const int bh    = blockIdx.y;        // 0..15
    const int b     = bh >> 3;
    const int h     = bh & 7;
    const int tid   = threadIdx.x;
    const int w     = tid >> 5;          // wave id 0..7
    const int lane  = tid & 31;
    const int l15   = lane & 15;
    const int hiA   = (lane >> 4) << 3;  // 0 / 8  : A-frag + C/D row offset
    const int hiB   = (lane >> 4) << 4;  // 0 / 16 : unused; B-frag K offset is hiA

    __shared__ float smem[64 * 64];
    #pragma unroll
    for (int t = 0; t < 16; ++t) smem[tid + 256 * t] = 0.0f;
    __syncthreads();

    const float* qb = q + (size_t)b * (SEQ * ROWSTRIDE) + h * DHEAD;
    const float* kb = k + (size_t)b * (SEQ * ROWSTRIDE) + h * DHEAD;

    float8 acc[4][4];
    #pragma unroll
    for (int m = 0; m < 4; ++m)
        #pragma unroll
        for (int n = 0; n < 4; ++n)
            acc[m][n] = (float8)0.0f;

    const int sbase = chunk * SCHUNK + w * SSUB;

    for (int ks = 0; ks < KSTEPS; ++ks) {
        const int sb = sbase + ks * 32;

        // B fragments: 32x16 f16, same K order as A:
        // element e of the v16h is K = e + (e>=8 ? 8 : 0) + hiA, N = n*16 + l15.
        half16 bf[4];
        #pragma unroll
        for (int n = 0; n < 4; ++n) {
            #pragma unroll
            for (int e = 0; e < 16; ++e) {
                const int kk = sb + hiA + e + ((e >= 8) ? 8 : 0);
                bf[n][e] = (_Float16)kb[kk * ROWSTRIDE + n * 16 + l15];
            }
        }

        // A fragments: 16x32 f16, A[M,K] = q[sb+K, m*16+M] (transposed slab).
        // Element e: K = e + (e>=8 ? 8 : 0) + hiA, M = l15.
        #pragma unroll
        for (int m = 0; m < 4; ++m) {
            half16 af;
            #pragma unroll
            for (int e = 0; e < 16; ++e) {
                const int kk = sb + hiA + e + ((e >= 8) ? 8 : 0);
                af[e] = (_Float16)qb[kk * ROWSTRIDE + m * 16 + l15];
            }
            #pragma unroll
            for (int n = 0; n < 4; ++n) {
                acc[m][n] = __builtin_amdgcn_wmma_f32_16x16x32_f16(
                    /*neg_a=*/false, af, /*neg_b=*/false, bf[n],
                    /*c_mod=*/(short)0, acc[m][n],
                    /*reuse_a=*/false, /*reuse_b=*/false);
            }
        }
    }

    // C/D layout: vgpr r holds M = m*16 + r + (lane>=16 ? 8 : 0), N = n*16 + l15.
    #pragma unroll
    for (int m = 0; m < 4; ++m)
        #pragma unroll
        for (int n = 0; n < 4; ++n)
            #pragma unroll
            for (int r = 0; r < 8; ++r) {
                const int M = m * 16 + r + hiA;
                const int N = n * 16 + l15;
                atomicAdd(&smem[M * 64 + N], acc[m][n][r]);   // ds_add_f32
            }
    __syncthreads();

    // One coalesced flush of the workgroup partial into global scores.
    float* wsbh = ws + bh * 4096;
    #pragma unroll
    for (int t = 0; t < 16; ++t) {
        const int idx = tid + 256 * t;
        unsafeAtomicAdd(&wsbh[idx], smem[idx]);               // global_atomic_add_f32
    }
}

// One wave (32 lanes) per 64-wide row: scale, max, exp, sum, normalize.
__global__ __launch_bounds__(32)
void softmax_rows(const float* __restrict__ ws, float* __restrict__ out)
{
    const int row  = blockIdx.x;          // 0..1023  (bh*64 + i)
    const int l    = threadIdx.x;
    const float sc = 0.125f;              // 64^-0.5

    float x0 = ws[row * 64 + l]      * sc;
    float x1 = ws[row * 64 + 32 + l] * sc;

    float m = fmaxf(x0, x1);
    #pragma unroll
    for (int off = 16; off > 0; off >>= 1)
        m = fmaxf(m, __shfl_xor(m, off, 32));

    float e0 = __expf(x0 - m);
    float e1 = __expf(x1 - m);
    float s  = e0 + e1;
    #pragma unroll
    for (int off = 16; off > 0; off >>= 1)
        s += __shfl_xor(s, off, 32);

    const float inv = 1.0f / s;
    out[row * 64 + l]      = e0 * inv;
    out[row * 64 + 32 + l] = e1 * inv;
}

extern "C" void kernel_launch(void* const* d_in, const int* in_sizes, int n_in,
                              void* d_out, int out_size, void* d_ws, size_t ws_size,
                              hipStream_t stream)
{
    const float* q = (const float*)d_in[0];
    const float* k = (const float*)d_in[1];
    // d_in[2] (value) is dead in the reference under jit DCE.
    float* ws  = (float*)d_ws;    // 16*64*64 fp32 = 256 KB of scratch
    float* out = (float*)d_out;

    hipMemsetAsync(d_ws, 0, BH * 64 * 64 * sizeof(float), stream);
    attn_scores_wmma<<<dim3(NCHUNK, BH), 256, 0, stream>>>(q, k, ws);
    softmax_rows<<<BH * 64, 32, 0, stream>>>(ws, out);
}
